// RacingMultiHeadAttention_88931592831332
// MI455X (gfx1250) — matrix operand order repaired and checked by reading, not compiled, not run on hardware
//
#include <hip/hip_runtime.h>
#include <hip/hip_bf16.h>
#include <math.h>

// ---------------------------------------------------------------------------
// RacingMultiHeadAttention for MI455X (gfx1250), wave32, WMMA f16 (f32 accum)
// B=2, S=2048, D=1024, H=16, dk=64
// GEMMs: TDM (tensor_load_to_lds, TENSORcnt) double-buffered tiles.
// Attention: per-lane async global->LDS (ASYNCcnt) double-buffered tiles.
// ---------------------------------------------------------------------------

typedef __attribute__((ext_vector_type(16))) _Float16 v16h;
typedef __attribute__((ext_vector_type(8)))  _Float16 v8h;
typedef __attribute__((ext_vector_type(8)))  float    v8f;
typedef unsigned int u32x4 __attribute__((ext_vector_type(4)));
typedef int          i32x4 __attribute__((ext_vector_type(4)));
typedef int          i32x8 __attribute__((ext_vector_type(8)));

#define B_SZ   2
#define S_SZ   2048
#define D_SZ   1024
#define H_SZ   16
#define DK_SZ  64
#define NTOK   (B_SZ * S_SZ)          // 4096

__device__ __forceinline__ v16h cat8(v8h lo, v8h hi) {
  return __builtin_shufflevector(lo, hi, 0,1,2,3,4,5,6,7,8,9,10,11,12,13,14,15);
}

__device__ __forceinline__ v8f wmma16(v16h a, v16h b, v8f c) {
  // D = A(16x32 f16) * B(32x16 f16) + C(16x16 f32)
  return __builtin_amdgcn_wmma_f32_16x16x32_f16(false, a, false, b,
                                                (short)0, c, false, false);
}

// ---- LDS offset of a __shared__ object (generic ptr low 32 bits) ----------
__device__ __forceinline__ unsigned lds_off(const void* p) {
  return (unsigned)(unsigned long long)p;
}

// ---- CDNA5 async global->LDS DMA (tracked by ASYNCcnt) --------------------
__device__ __forceinline__ void async_b128(unsigned ldsoff, const _Float16* g) {
  unsigned long long ga = (unsigned long long)g;
  asm volatile("global_load_async_to_lds_b128 %0, %1, off"
               :: "v"(ldsoff), "v"(ga) : "memory");
}
#define WAIT_ASYNC_0() asm volatile("s_wait_asynccnt 0x0" ::: "memory")
#define WAIT_ASYNC_4() asm volatile("s_wait_asynccnt 0x4" ::: "memory")

// ---- CDNA5 Tensor Data Mover: 2D tile (f16) global -> LDS -----------------
// D# per CDNA5 ISA: group0 = {count=1, lds_addr, global_addr[56:0], type=2}
// group1 = {data_size=2B, tensor_dim0/1, tile_dim0/1, tensor_dim0_stride}
__device__ __forceinline__ void tdm_load_2d_f16(unsigned ldsoff, const void* gaddr,
                                                unsigned tile_x, unsigned tile_y,
                                                unsigned tensor_x, unsigned tensor_y,
                                                unsigned stride_x) {
  unsigned long long ga = (unsigned long long)gaddr;
  u32x4 g0;
  g0[0] = 1u;                                      // count=1 (valid user D#)
  g0[1] = ldsoff;                                  // lds_addr (bytes)
  g0[2] = (unsigned)ga;                            // global_addr[31:0]
  g0[3] = (unsigned)(ga >> 32) | (2u << 30);       // global_addr[56:32] | type=2
  i32x8 g1;
  g1[0] = 1 << 16;                                 // data_size=1 -> 2 bytes
  g1[1] = (int)((tensor_x & 0xFFFFu) << 16);       // tensor_dim0[15:0] @63:48
  g1[2] = (int)((tensor_x >> 16) | ((tensor_y & 0xFFFFu) << 16)); // dim0 hi|dim1 lo
  g1[3] = (int)((tensor_y >> 16) | (tile_x << 16));// dim1 hi | tile_dim0
  g1[4] = (int)tile_y;                             // tile_dim1 (tile_dim2=0)
  g1[5] = (int)stride_x;                           // tensor_dim0_stride[31:0]
  g1[6] = 0;
  g1[7] = 0;
  i32x4 z = {0, 0, 0, 0};
#if defined(__clang_major__) && (__clang_major__ >= 23)
  i32x8 z8 = {0, 0, 0, 0, 0, 0, 0, 0};
  __builtin_amdgcn_tensor_load_to_lds(g0, g1, z, z, z8, 0);
#else
  __builtin_amdgcn_tensor_load_to_lds(g0, g1, z, z, 0);
#endif
}
#define WAIT_TENSOR(n) __builtin_amdgcn_s_wait_tensorcnt((short)(n))

// ------------------------------ f32 -> f16 ---------------------------------
__global__ void cvt_f32_to_f16(const float* __restrict__ in,
                               _Float16* __restrict__ out, int n) {
  int i = blockIdx.x * blockDim.x + threadIdx.x;
  if (i < n) out[i] = (_Float16)in[i];
}

// ----------------- V transpose: [B,S,H*dk] -> [B,H,dk,S] -------------------
__global__ __launch_bounds__(256)
void transpose_v(const _Float16* __restrict__ Vp, _Float16* __restrict__ Vt) {
  __shared__ _Float16 t[64][65];      // padded to dodge bank conflicts
  const int bh = blockIdx.y, b = bh >> 4, h = bh & 15;
  const int s0 = blockIdx.x * 64;
  const int tid = threadIdx.x;
  const int r = tid >> 2;             // 0..63
  const int c = (tid & 3) * 16;       // 0,16,32,48
  const _Float16* src = &Vp[((size_t)b * S_SZ + s0 + r) * D_SZ + h * DK_SZ + c];
  v8h a0 = *(const v8h*)src;
  v8h a1 = *(const v8h*)(src + 8);
  #pragma unroll
  for (int i = 0; i < 8; ++i) { t[c + i][r] = a0[i]; t[c + 8 + i][r] = a1[i]; }
  __syncthreads();
  _Float16* dst = &Vt[(((size_t)b * H_SZ + h) * DK_SZ + r) * S_SZ + s0 + c];
  v8h o0, o1;
  #pragma unroll
  for (int i = 0; i < 8; ++i) { o0[i] = t[r][c + i]; o1[i] = t[r][c + 8 + i]; }
  *(v8h*)dst = o0;
  *(v8h*)(dst + 8) = o1;
}

// ------------------------- tiled WMMA GEMM ---------------------------------
// C[M=4096][N=1024] = A[4096][K=1024] @ W^T + bias, where B(k,n) = Wt[n][k].
// Block tile 128x128, BK=64; 8 waves in 2(M) x 4(N); wave tile 64x32.
// Tiles staged by the Tensor Data Mover, double buffered (TENSORcnt).
template <bool OUT_F16>
__global__ __launch_bounds__(256)
void gemm_wmma(const _Float16* __restrict__ A,
               const _Float16* __restrict__ Wt,
               const float*    __restrict__ bias,
               void*           __restrict__ Cout) {
  constexpr int KD = 1024, ND = 1024, BK = 64;
  __shared__ _Float16 As[2][128 * BK];   // [buf][row][k]  2 x 16KB
  __shared__ _Float16 Bs[2][128 * BK];   // [buf][n][k]    2 x 16KB

  const int tid  = threadIdx.x;
  const int wid  = tid >> 5;
  const int lane = tid & 31;
  const int half = lane >> 4;
  const int l16  = lane & 15;

  const int m0 = blockIdx.x * 128;
  const int n0 = blockIdx.y * 128;
  const int wm = (wid >> 2) * 64;     // 0 or 64
  const int wn = (wid & 3) * 32;      // 0,32,64,96

  v8f acc[4][2];
  #pragma unroll
  for (int mt = 0; mt < 4; ++mt)
    #pragma unroll
    for (int nt = 0; nt < 2; ++nt) acc[mt][nt] = (v8f)0.0f;

  auto issue_ab = [&](int buf, int k0) {
    if (wid == 0) {   // one wave drives the TDM (EXEC ignored by tensor ops)
      tdm_load_2d_f16(lds_off(&As[buf][0]), &A[(size_t)m0 * KD + k0],
                      BK, 128, KD, NTOK, KD);
      tdm_load_2d_f16(lds_off(&Bs[buf][0]), &Wt[(size_t)n0 * KD + k0],
                      BK, 128, KD, ND, KD);
    }
  };

  issue_ab(0, 0);
  for (int it = 0; it < KD / BK; ++it) {   // 16 k-stages
    const int cur = it & 1;
    if (it + 1 < KD / BK) {
      issue_ab(1 - cur, (it + 1) * BK);
      WAIT_TENSOR(2);                  // current tile done, prefetch in flight
    } else {
      WAIT_TENSOR(0);
    }
    __syncthreads();

    #pragma unroll
    for (int ks = 0; ks < 2; ++ks) {
      // B fragments: lane L -> col n = l16, K runs half*8..+7 and 16+half*8..+7
      v16h bfr[2];
      #pragma unroll
      for (int nt = 0; nt < 2; ++nt) {
        const _Float16* p = &Bs[cur][(wn + nt * 16 + l16) * BK + ks * 32 + half * 8];
        bfr[nt] = cat8(*(const v8h*)p, *(const v8h*)(p + 16));
      }
      // A fragments: lane L -> row m = l16, K runs half*8..+7 and 16+half*8..+7
      #pragma unroll
      for (int mt = 0; mt < 4; ++mt) {
        const _Float16* p = &As[cur][(wm + mt * 16 + l16) * BK + ks * 32 + half * 8];
        v16h afr = cat8(*(const v8h*)p, *(const v8h*)(p + 16));
        acc[mt][0] = wmma16(afr, bfr[0], acc[mt][0]);
        acc[mt][1] = wmma16(afr, bfr[1], acc[mt][1]);
      }
    }
    __syncthreads();
  }

  // epilogue: C element (m,n): lane = n%16 + 16*(m/8 within tile), vgpr = m%8
  #pragma unroll
  for (int mt = 0; mt < 4; ++mt) {
    #pragma unroll
    for (int nt = 0; nt < 2; ++nt) {
      #pragma unroll
      for (int i = 0; i < 8; ++i) {
        int row = m0 + wm + mt * 16 + half * 8 + i;
        int col = n0 + wn + nt * 16 + l16;
        float v = acc[mt][nt][i] + bias[col];
        if (OUT_F16)
          ((_Float16*)Cout)[(size_t)row * ND + col] = (_Float16)v;
        else
          ((float*)Cout)[(size_t)row * ND + col] = v;
      }
    }
  }
}

// --------------------------- flash attention -------------------------------
// grid: (S/128, B*H). 8 waves; wave owns 16 q-rows, full dk=64 output.
// K and V^T tiles double-buffered via async global->LDS (ASYNCcnt).
__global__ __launch_bounds__(256)
void attn_wmma(const _Float16* __restrict__ Q,    // [B,S,1024] col = h*64+d
               const _Float16* __restrict__ K,    // [B,S,1024]
               const _Float16* __restrict__ Vt,   // [B,H,64,S]
               const float*    __restrict__ rbias,
               _Float16*       __restrict__ ctx) {
  constexpr int BQ = 128, BKV = 64;
  __shared__ _Float16 Ks[2][BKV * DK_SZ];   // [buf][kv][d]  2 x 8KB
  __shared__ _Float16 Vs[2][DK_SZ * BKV];   // [buf][d][kv]  2 x 8KB
  __shared__ _Float16 Ps[8 * 16 * BKV];     // per-wave P tiles, 16KB

  const int bh  = blockIdx.y;
  const int b   = bh >> 4;
  const int h   = bh & 15;
  const int q0  = blockIdx.x * BQ;
  const int tid = threadIdx.x;
  const int wid = tid >> 5;
  const int lane = tid & 31;
  const int half = lane >> 4;
  const int l16  = lane & 15;

  const size_t base  = (size_t)b * S_SZ * D_SZ + (size_t)h * DK_SZ;
  const _Float16* Vtb = Vt + ((size_t)b * H_SZ + h) * DK_SZ * S_SZ;
  const float hb     = rbias[h];
  const float scale  = 0.125f;              // 1/sqrt(64)
  const float log2e  = 1.44269504088896f;

  // Q A-fragments for this wave's 16 rows (registers for whole kernel)
  v16h qf[2];
  {
    const _Float16* qp = &Q[base + (size_t)(q0 + wid * 16 + l16) * D_SZ];
    #pragma unroll
    for (int ks = 0; ks < 2; ++ks) {
      v8h lo = *(const v8h*)&qp[ks * 32 + half * 8];
      v8h hi = *(const v8h*)&qp[ks * 32 + 16 + half * 8];
      qf[ks] = cat8(lo, hi);
    }
  }

  v8f o[4];
  #pragma unroll
  for (int nt = 0; nt < 4; ++nt) o[nt] = (v8f)0.0f;
  float mrow[8], lrow[8];
  #pragma unroll
  for (int i = 0; i < 8; ++i) { mrow[i] = -INFINITY; lrow[i] = 0.0f; }

  _Float16* Pw = &Ps[wid * 16 * BKV];

  auto issue_kv = [&](int buf, int kv0) {
    #pragma unroll
    for (int j = 0; j < 2; ++j) {
      const int cch = tid * 2 + j;      // 0..511 chunks of 16B per tile
      const int row = cch >> 3;         // 0..63
      const int off = (cch & 7) * 8;    // 0..56
      async_b128(lds_off(&Ks[buf][row * DK_SZ + off]),
                 &K[base + (size_t)(kv0 + row) * D_SZ + off]);
      async_b128(lds_off(&Vs[buf][row * BKV + off]),
                 &Vtb[(size_t)row * S_SZ + kv0 + off]);
    }
  };

  issue_kv(0, 0);
  for (int it = 0; it < S_SZ / BKV; ++it) {   // 32 kv tiles
    const int cur = it & 1;
    if (it + 1 < S_SZ / BKV) {
      issue_kv(1 - cur, (it + 1) * BKV);
      WAIT_ASYNC_4();
    } else {
      WAIT_ASYNC_0();
    }
    __syncthreads();

    // S = Q @ K^T : 4 n-tiles of 16 kv, 2 k-steps over dk
    v8f sf[4];
    #pragma unroll
    for (int nt = 0; nt < 4; ++nt) {
      v8f c = (v8f)0.0f;
      #pragma unroll
      for (int ks = 0; ks < 2; ++ks) {
        const _Float16* p = &Ks[cur][(nt * 16 + l16) * DK_SZ + ks * 32 + half * 8];
        c = wmma16(qf[ks], cat8(*(const v8h*)p, *(const v8h*)(p + 16)), c);
      }
      sf[nt] = c;
    }

    // scale + per-head bias, online softmax stats (rows = half*8+i)
    float mnew[8];
    #pragma unroll
    for (int i = 0; i < 8; ++i) mnew[i] = mrow[i];
    #pragma unroll
    for (int nt = 0; nt < 4; ++nt)
      #pragma unroll
      for (int i = 0; i < 8; ++i) {
        float v = sf[nt][i] * scale + hb;
        sf[nt][i] = v;
        mnew[i] = fmaxf(mnew[i], v);
      }
    #pragma unroll
    for (int off = 8; off > 0; off >>= 1)
      #pragma unroll
      for (int i = 0; i < 8; ++i)
        mnew[i] = fmaxf(mnew[i], __shfl_xor(mnew[i], off, 32));

    float alpha[8], rsum[8];
    #pragma unroll
    for (int i = 0; i < 8; ++i) {
      alpha[i] = exp2f((mrow[i] - mnew[i]) * log2e);
      mrow[i]  = mnew[i];
      rsum[i]  = 0.0f;
    }
    #pragma unroll
    for (int nt = 0; nt < 4; ++nt)
      #pragma unroll
      for (int i = 0; i < 8; ++i) o[nt][i] *= alpha[i];

    // P = exp(S - m), stash to wave-private LDS in [row][kv] layout
    #pragma unroll
    for (int nt = 0; nt < 4; ++nt)
      #pragma unroll
      for (int i = 0; i < 8; ++i) {
        float p = exp2f((sf[nt][i] - mrow[i]) * log2e);
        rsum[i] += p;
        Pw[(half * 8 + i) * BKV + nt * 16 + l16] = (_Float16)p;
      }
    #pragma unroll
    for (int off = 8; off > 0; off >>= 1)
      #pragma unroll
      for (int i = 0; i < 8; ++i) rsum[i] += __shfl_xor(rsum[i], off, 32);
    #pragma unroll
    for (int i = 0; i < 8; ++i) lrow[i] = lrow[i] * alpha[i] + rsum[i];

    // O += P @ V : A-frags of P from LDS, B-frags from transposed V tile
    #pragma unroll
    for (int ks = 0; ks < 2; ++ks) {
      const _Float16* pp = &Pw[l16 * BKV + ks * 32 + half * 8];
      v16h af = cat8(*(const v8h*)pp, *(const v8h*)(pp + 16));
      #pragma unroll
      for (int nt = 0; nt < 4; ++nt) {
        const _Float16* vb = &Vs[cur][(nt * 16 + l16) * BKV + ks * 32 + half * 8];
        o[nt] = wmma16(af, cat8(*(const v8h*)vb, *(const v8h*)(vb + 16)), o[nt]);
      }
    }
    __syncthreads();
  }

  // finalize: divide by row sums, write ctx in [B,S,H*dk] layout
  #pragma unroll
  for (int i = 0; i < 8; ++i) {
    float inv = 1.0f / lrow[i];
    int srow = q0 + wid * 16 + half * 8 + i;
    #pragma unroll
    for (int nt = 0; nt < 4; ++nt) {
      ctx[((size_t)b * S_SZ + srow) * D_SZ + h * DK_SZ + nt * 16 + l16] =
          (_Float16)(o[nt][i] * inv);
    }
  }
}

// ------------------------------- launcher ----------------------------------
extern "C" void kernel_launch(void* const* d_in, const int* in_sizes, int n_in,
                              void* d_out, int out_size, void* d_ws, size_t ws_size,
                              hipStream_t stream) {
  const float* query = (const float*)d_in[0];
  const float* key   = (const float*)d_in[1];
  const float* value = (const float*)d_in[2];
  const float* wq    = (const float*)d_in[3];
  const float* bq    = (const float*)d_in[4];
  const float* wk    = (const float*)d_in[5];
  const float* bk    = (const float*)d_in[6];
  const float* wv    = (const float*)d_in[7];
  const float* bv    = (const float*)d_in[8];
  const float* wo    = (const float*)d_in[9];
  const float* bo    = (const float*)d_in[10];
  const float* rbias = (const float*)d_in[11];

  const size_t nTok = (size_t)NTOK * D_SZ;   // 4M elements
  const size_t nW   = (size_t)D_SZ * D_SZ;   // 1M elements

  _Float16* p   = (_Float16*)d_ws;
  _Float16* qx  = p;  p += nTok;
  _Float16* kx  = p;  p += nTok;
  _Float16* vx  = p;  p += nTok;
  _Float16* wqh = p;  p += nW;
  _Float16* wkh = p;  p += nW;
  _Float16* wvh = p;  p += nW;
  _Float16* woh = p;  p += nW;
  _Float16* Qp  = p;  p += nTok;
  _Float16* Kp  = p;  p += nTok;
  _Float16* Vp  = p;  p += nTok;
  _Float16* Vtp = p;  p += nTok;
  _Float16* ctx = p;  p += nTok;

  // f32 -> f16 conversions
  cvt_f32_to_f16<<<nTok / 256, 256, 0, stream>>>(query, qx, (int)nTok);
  cvt_f32_to_f16<<<nTok / 256, 256, 0, stream>>>(key,   kx, (int)nTok);
  cvt_f32_to_f16<<<nTok / 256, 256, 0, stream>>>(value, vx, (int)nTok);
  cvt_f32_to_f16<<<nW / 256, 256, 0, stream>>>(wq, wqh, (int)nW);
  cvt_f32_to_f16<<<nW / 256, 256, 0, stream>>>(wk, wkh, (int)nW);
  cvt_f32_to_f16<<<nW / 256, 256, 0, stream>>>(wv, wvh, (int)nW);
  cvt_f32_to_f16<<<nW / 256, 256, 0, stream>>>(wo, woh, (int)nW);

  // Q/K/V projections (f16 out)
  dim3 ggrid(NTOK / 128, D_SZ / 128);
  gemm_wmma<true><<<ggrid, 256, 0, stream>>>(qx, wqh, bq, Qp);
  gemm_wmma<true><<<ggrid, 256, 0, stream>>>(kx, wkh, bk, Kp);
  gemm_wmma<true><<<ggrid, 256, 0, stream>>>(vx, wvh, bv, Vp);

  // V transpose to [B,H,dk,S]
  transpose_v<<<dim3(S_SZ / 64, B_SZ * H_SZ), 256, 0, stream>>>(Vp, Vtp);

  // flash attention
  attn_wmma<<<dim3(S_SZ / 128, B_SZ * H_SZ), 256, 0, stream>>>(Qp, Kp, Vtp, rbias, ctx);

  // output projection (f32 out)
  gemm_wmma<false><<<ggrid, 256, 0, stream>>>(ctx, woh, bo, d_out);
}
